// MHBNN_13898514170060
// MI455X (gfx1250) — matrix operand order repaired and checked by reading, not compiled, not run on hardware
//
#include <hip/hip_runtime.h>

#define DEV __device__ __forceinline__

typedef __attribute__((ext_vector_type(16))) _Float16 v16h;
typedef __attribute__((ext_vector_type(16))) __bf16   v16bf;
typedef __attribute__((ext_vector_type(8)))  float    v8f;

// Problem constants
#define NI   64
#define NC   512
#define NH   32
#define NW   32
#define NN   256
#define NV   8
#define RR   7
#define ND   128
#define NS   392            // V*R*R real samples
#define SPAD 416            // padded sample dim (13*32)
#define NF   512
#define DD   (ND*ND)        // 16384
#define HWSZ (NH*NW)

// ---------------- WMMA fragment loaders (CDNA5 wave32 layouts) ----------------
// A (16x32, 16-bit): lane m = lane&15; kb = (lane>>4)*8; halves 0..7 -> K=kb+h,
// halves 8..15 -> K=kb+16+h.
template <typename ET>
DEV v16h load_a_h(const ET* base, int mstride, int kstride) {
  int lane = threadIdx.x & 31;
  const ET* row = base + (lane & 15) * mstride;
  int kb = (lane >> 4) * 8;
  v16h a;
#pragma unroll
  for (int h = 0; h < 8; ++h) {
    a[h]     = (_Float16)row[(kb + h) * kstride];
    a[h + 8] = (_Float16)row[(kb + 16 + h) * kstride];
  }
  return a;
}
template <typename ET>
DEV v16bf load_a_bf(const ET* base, int mstride, int kstride) {
  int lane = threadIdx.x & 31;
  const ET* row = base + (lane & 15) * mstride;
  int kb = (lane >> 4) * 8;
  v16bf a;
#pragma unroll
  for (int h = 0; h < 8; ++h) {
    a[h]     = (__bf16)row[(kb + h) * kstride];
    a[h + 8] = (__bf16)row[(kb + 16 + h) * kstride];
  }
  return a;
}
// B (32x16, 16-bit): lane col = lane&15; kb = (lane>>4)*8; halves 0..7 -> K=kb+h,
// halves 8..15 -> K=kb+16+h.
// element [k][col] at base[k*kstride + col*cstride]
template <typename ET>
DEV v16h load_b_h(const ET* base, int kstride, int cstride) {
  int lane = threadIdx.x & 31;
  const ET* p = base + (lane & 15) * cstride;
  int kb = (lane >> 4) * 8;
  v16h b;
#pragma unroll
  for (int h = 0; h < 16; ++h) b[h] = (_Float16)p[(kb + (h & 7) + (h >> 3) * 16) * kstride];
  return b;
}
template <typename ET>
DEV v16bf load_b_bf(const ET* base, int kstride, int cstride) {
  int lane = threadIdx.x & 31;
  const ET* p = base + (lane & 15) * cstride;
  int kb = (lane >> 4) * 8;
  v16bf b;
#pragma unroll
  for (int h = 0; h < 16; ++h) b[h] = (__bf16)p[(kb + (h & 7) + (h >> 3) * 16) * kstride];
  return b;
}

DEV v8f wmma_h(v16h a, v16h b, v8f c) {
  return __builtin_amdgcn_wmma_f32_16x16x32_f16(false, a, false, b, (short)0, c,
                                                false, false);
}
DEV v8f wmma_bf(v16bf a, v16bf b, v8f c) {
  return __builtin_amdgcn_wmma_f32_16x16x32_bf16(false, a, false, b, (short)0, c,
                                                 false, false);
}

DEV float sign_sqrtf(float x) {
  float sg = (x > 0.f) ? 1.f : ((x < 0.f) ? -1.f : 0.f);
  return sg * sqrtf(fabsf(x) + 1e-12f);
}

// ---------------- Kernel 0: f32 -> f16 convert ----------------
__global__ void k_cvt16(const float* __restrict__ src, _Float16* __restrict__ dst,
                        int nElem) {
  int i = blockIdx.x * blockDim.x + threadIdx.x;
  if (i < nElem) dst[i] = (_Float16)src[i];
}

// ---------------- Kernel 1: ROI-align + sign_sqrt + projection GEMM ----------
// grid (26, 256): blockIdx.x = s-tile (16 samples of SPAD), blockIdx.y = n.
// 8 waves; wave w owns d-rows [16w,16w+16). m_out: f16 [N][128][416].
__global__ void __launch_bounds__(256) k_roi_project(
    const float* __restrict__ images, const float* __restrict__ bboxes,
    const int* __restrict__ kf, const _Float16* __restrict__ pw16,
    const float* __restrict__ proj_b, _Float16* __restrict__ mOut) {
  const int stile = blockIdx.x;
  const int n = blockIdx.y;
  const int tid = threadIdx.x;
  const int lane = tid & 31;
  const int wv = tid >> 5;

  __shared__ int      s_ibase[16];
  __shared__ int      s_off[16][4];
  __shared__ float    s_w[16][4];
  __shared__ _Float16 roiB[32][18];   // [k][sample], padded

  if (tid < 16) {
    int s = stile * 16 + tid;
    if (s < NS) {
      int v = s / 49; int rem = s - v * 49;
      int ry = rem / 7; int rx = rem - ry * 7;
      const float* bb = bboxes + ((n * NV + v) << 2);
      float bx0 = bb[0], by0 = bb[1], bx1 = bb[2], by1 = bb[3];
      float x0 = fminf(bx0, bx1) * (float)NW;
      float x1 = (fmaxf(bx0, bx1) + 0.001f) * (float)NW;
      float y0 = fminf(by0, by1) * (float)NH;
      float y1 = (fmaxf(by0, by1) + 0.001f) * (float)NH;
      float gx = ((float)rx + 0.5f) / (float)RR;
      float gy = ((float)ry + 0.5f) / (float)RR;
      float xs = fminf(fmaxf(x0 + gx * (x1 - x0), 0.f), (float)(NW - 1));
      float ys = fminf(fmaxf(y0 + gy * (y1 - y0), 0.f), (float)(NH - 1));
      int xlo = (int)floorf(xs); int xhi = min(xlo + 1, NW - 1);
      int ylo = (int)floorf(ys); int yhi = min(ylo + 1, NH - 1);
      float wx = xs - (float)xlo, wy = ys - (float)ylo;
      int img = kf[n * NV + v];
      s_ibase[tid]  = img * (NC * HWSZ);
      s_off[tid][0] = ylo * NW + xlo;
      s_off[tid][1] = ylo * NW + xhi;
      s_off[tid][2] = yhi * NW + xlo;
      s_off[tid][3] = yhi * NW + xhi;
      s_w[tid][0] = (1.f - wy) * (1.f - wx);
      s_w[tid][1] = (1.f - wy) * wx;
      s_w[tid][2] = wy * (1.f - wx);
      s_w[tid][3] = wy * wx;
    } else {
      s_ibase[tid] = 0;
      s_off[tid][0] = s_off[tid][1] = s_off[tid][2] = s_off[tid][3] = 0;
      s_w[tid][0] = s_w[tid][1] = s_w[tid][2] = s_w[tid][3] = 0.f;
    }
  }
  __syncthreads();

  v8f acc = {};
  for (int kc = 0; kc < NC / 32; ++kc) {
    // gather a 32(ch) x 16(sample) chunk; 512 elems, 2 per thread
#pragma unroll
    for (int e = tid; e < 512; e += 256) {
      int k = e >> 4, sl = e & 15;
      int c = kc * 32 + k;
      const float* img = images + s_ibase[sl] + c * HWSZ;
      float val = s_w[sl][0] * img[s_off[sl][0]] +
                  s_w[sl][1] * img[s_off[sl][1]] +
                  s_w[sl][2] * img[s_off[sl][2]] +
                  s_w[sl][3] * img[s_off[sl][3]];
      roiB[k][sl] = (_Float16)sign_sqrtf(val);
    }
    __syncthreads();
    v16h a = load_a_h(pw16 + wv * 16 * NC + kc * 32, NC, 1);
    v16h b = load_b_h(&roiB[0][0], 18, 1);
    acc = wmma_h(a, b, acc);
    __syncthreads();
  }

  int ncol = lane & 15;
  int mo = (lane >> 4) * 8;
  int scol = stile * 16 + ncol;
  _Float16* dst = mOut + ((long)n * ND + wv * 16) * SPAD + stile * 16 + ncol;
#pragma unroll
  for (int r = 0; r < 8; ++r) {
    float val = acc[r] + proj_b[wv * 16 + mo + r];
    dst[(mo + r) * SPAD] = (scol < NS) ? (_Float16)val : (_Float16)0.f;
  }
}

// ---------------- Kernel 2: Gram B = m * m^T ----------------
__global__ void __launch_bounds__(256) k_gram(const _Float16* __restrict__ m,
                                              float* __restrict__ B) {
  const int n = blockIdx.x;
  const int wv = threadIdx.x >> 5;
  const _Float16* mn = m + (long)n * ND * SPAD;
  v8f acc[8] = {};
  for (int kc = 0; kc < SPAD / 32; ++kc) {
    if (kc + 4 < SPAD / 32) {
      __builtin_prefetch(mn + wv * 16 * SPAD + (kc + 4) * 32, 0, 0);
    }
    v16h a = load_a_h(mn + wv * 16 * SPAD + kc * 32, SPAD, 1);
#pragma unroll
    for (int ct = 0; ct < 8; ++ct) {
      v16h b = load_b_h(mn + ct * 16 * SPAD + kc * 32, 1, SPAD);
      acc[ct] = wmma_h(a, b, acc[ct]);
    }
  }
  int lane = threadIdx.x & 31;
  int ncol = lane & 15, mo = (lane >> 4) * 8;
  float* Bn = B + ((long)n * ND + wv * 16) * ND;
#pragma unroll
  for (int ct = 0; ct < 8; ++ct)
#pragma unroll
    for (int r = 0; r < 8; ++r)
      Bn[(mo + r) * ND + ct * 16 + ncol] = acc[ct][r];
}

// ---------------- Kernel 3: parallel cyclic Jacobi eigensolver ---------------
// one block of 256 threads (8 waves) per matrix. A,V in LDS (132KB of 320KB).
// Each Jacobi step applies 64 disjoint Givens rotations; the (row x pair) /
// (col x pair) update grids are split across two 128-thread halves.
__global__ void __launch_bounds__(256) k_jacobi(const float* __restrict__ B,
                                                const float* __restrict__ lambdas,
                                                __bf16* __restrict__ U,
                                                __bf16* __restrict__ Wout) {
  const int n = blockIdx.x;
  const int tid = threadIdx.x;        // 0..255
  const int row = tid & 127;
  const int half = tid >> 7;          // 0 or 1
  __shared__ float As[ND * 129];
  __shared__ float Vs[ND * 129];
  __shared__ float s_c[64], s_s[64];
  __shared__ int   s_p[64], s_q[64];
  __shared__ float ev[ND], hsv[ND];
  __shared__ int   ord[ND];

  const float* Bn = B + (long)n * DD;
  for (int col = half * 64; col < half * 64 + 64; ++col) {
    As[row * 129 + col] = Bn[row * ND + col];
    Vs[row * 129 + col] = (row == col) ? 1.f : 0.f;
  }
  __syncthreads();

  for (int sw = 0; sw < 10; ++sw) {
    for (int t = 0; t < 127; ++t) {
      if (tid < 64) {
        // round-robin tournament pairing
        int p = (tid == 0) ? 0 : 1 + (tid - 1 + t) % 127;
        int q = 1 + (127 - tid - 1 + t) % 127;
        if (p > q) { int tmp = p; p = q; q = tmp; }
        float app = As[p * 129 + p], aqq = As[q * 129 + q], apq = As[p * 129 + q];
        float c = 1.f, s = 0.f;
        if (fabsf(apq) > 1e-30f) {
          float tau = (aqq - app) / (2.f * apq);
          float tt = copysignf(1.f, tau) / (fabsf(tau) + sqrtf(1.f + tau * tau));
          c = rsqrtf(1.f + tt * tt);
          s = tt * c;
        }
        s_c[tid] = c; s_s[tid] = s; s_p[tid] = p; s_q[tid] = q;
      }
      __syncthreads();
      // column update A <- A*G, V <- V*G  (thread owns row; half the pairs)
      {
        float* Ar = As + row * 129;
        float* Vr = Vs + row * 129;
        for (int j = half * 32; j < half * 32 + 32; ++j) {
          int p = s_p[j], q = s_q[j];
          float c = s_c[j], s = s_s[j];
          float x = Ar[p], y = Ar[q];
          Ar[p] = c * x - s * y; Ar[q] = s * x + c * y;
          x = Vr[p]; y = Vr[q];
          Vr[p] = c * x - s * y; Vr[q] = s * x + c * y;
        }
      }
      __syncthreads();
      // row update A <- G^T*A  (thread owns column; half the pairs)
      for (int j = half * 32; j < half * 32 + 32; ++j) {
        int p = s_p[j], q = s_q[j];
        float c = s_c[j], s = s_s[j];
        float x = As[p * 129 + row], y = As[q * 129 + row];
        As[p * 129 + row] = c * x - s * y;
        As[q * 129 + row] = s * x + c * y;
      }
      __syncthreads();
    }
  }

  if (tid < ND) ev[tid] = As[tid * 129 + tid];
  __syncthreads();
  if (tid < ND) {
    float mv = ev[tid];
    int rank = 0;
    for (int j = 0; j < ND; ++j) {
      float o = ev[j];
      rank += (o > mv) || (o == mv && j < tid) ? 1 : 0;
    }
    ord[rank] = tid;
  }
  __syncthreads();
  if (tid < ND) {
    float S = fmaxf(ev[ord[tid]], 1e-12f);   // EPS clamp (PSD spectrum)
    float lam = lambdas[tid];
    hsv[tid] = (fabsf(lam) > 1e-12f) ? (powf(S, lam) - 1.f) / lam : logf(S);
  }
  __syncthreads();
  __bf16* Un = U + (long)n * DD + row * ND;
  __bf16* Wn = Wout + (long)n * DD + row * ND;
  for (int r = half * 64; r < half * 64 + 64; ++r) {
    float u = Vs[row * 129 + ord[r]];
    Un[r] = (__bf16)u;
    Wn[r] = (__bf16)(u * hsv[r]);
  }
}

// ---------------- Kernel 4: Brec = W * U^T  (bf16 WMMA) ----------------
__global__ void __launch_bounds__(256) k_recon(const __bf16* __restrict__ Wm,
                                               const __bf16* __restrict__ Um,
                                               float* __restrict__ Brec) {
  const int n = blockIdx.x;
  const int wv = threadIdx.x >> 5;
  const __bf16* Wn = Wm + (long)n * DD;
  const __bf16* Un = Um + (long)n * DD;
  v8f acc[8] = {};
  for (int kc = 0; kc < ND / 32; ++kc) {
    v16bf a = load_a_bf(Wn + wv * 16 * ND + kc * 32, ND, 1);
#pragma unroll
    for (int ct = 0; ct < 8; ++ct) {
      // (U^T)[k][col] = U[col][k]
      v16bf b = load_b_bf(Un + ct * 16 * ND + kc * 32, 1, ND);
      acc[ct] = wmma_bf(a, b, acc[ct]);
    }
  }
  int lane = threadIdx.x & 31;
  int ncol = lane & 15, mo = (lane >> 4) * 8;
  float* Bn = Brec + ((long)n * ND + wv * 16) * ND;
#pragma unroll
  for (int ct = 0; ct < 8; ++ct)
#pragma unroll
    for (int r = 0; r < 8; ++r)
      Bn[(mo + r) * ND + ct * 16 + ncol] = acc[ct][r];
}

// ---------------- Kernel 5: sign_sqrt + L2 normalize -> flat f16 -------------
__global__ void __launch_bounds__(256) k_normflat(const float* __restrict__ Brec,
                                                  _Float16* __restrict__ flat) {
  const int n = blockIdx.x;
  const int tid = threadIdx.x;
  __shared__ float red[256];
  const float* Br = Brec + (long)n * DD;
  float part = 0.f;
  for (int i = tid; i < DD; i += 256) part += fabsf(Br[i]) + 1e-12f;  // ss(x)^2
  red[tid] = part;
  __syncthreads();
  for (int sft = 128; sft > 0; sft >>= 1) {
    if (tid < sft) red[tid] += red[tid + sft];
    __syncthreads();
  }
  float inv = 1.f / (sqrtf(red[0]) + 1e-8f);
  _Float16* fn = flat + (long)n * DD;
  for (int i = tid; i < DD; i += 256)
    fn[i] = (_Float16)(sign_sqrtf(Br[i]) * inv);
}

// ---------------- Kernel 6: out = flat @ proj2_w^T + b ----------------
// grid (4,16): blockIdx.y = m-tile (256/16), ntile = blockIdx.x*8 + wave.
__global__ void __launch_bounds__(256) k_out(const _Float16* __restrict__ flat,
                                             const _Float16* __restrict__ p2,
                                             const float* __restrict__ p2b,
                                             float* __restrict__ out) {
  const int mtile = blockIdx.y;
  const int ntile = blockIdx.x * 8 + (threadIdx.x >> 5);
  const _Float16* A = flat + (long)mtile * 16 * DD;
  const _Float16* Bp = p2 + (long)ntile * 16 * DD;
  v8f acc = {};
  for (int kc = 0; kc < DD / 32; ++kc) {
    if (kc + 8 < DD / 32) {
      // stream-ahead into cache hierarchy (global_prefetch_b8)
      __builtin_prefetch(A + (kc + 8) * 32, 0, 0);
      __builtin_prefetch(Bp + (kc + 8) * 32, 0, 0);
    }
    v16h a = load_a_h(A + kc * 32, DD, 1);
    v16h b = load_b_h(Bp + kc * 32, 1, DD);   // B[k][f] = p2[f][k]
    acc = wmma_h(a, b, acc);
  }
  int lane = threadIdx.x & 31;
  int ncol = lane & 15, mo = (lane >> 4) * 8;
  float bias = p2b[ntile * 16 + ncol];
#pragma unroll
  for (int r = 0; r < 8; ++r)
    out[(long)(mtile * 16 + mo + r) * NF + ntile * 16 + ncol] = acc[r] + bias;
}

// ---------------- launcher ----------------
extern "C" void kernel_launch(void* const* d_in, const int* in_sizes, int n_in,
                              void* d_out, int out_size, void* d_ws, size_t ws_size,
                              hipStream_t stream) {
  const float* images  = (const float*)d_in[0];
  const float* bboxes  = (const float*)d_in[1];
  const int*   kf      = (const int*)d_in[2];
  const float* proj_w  = (const float*)d_in[3];
  const float* proj_b  = (const float*)d_in[4];
  const float* proj2_w = (const float*)d_in[5];
  const float* proj2_b = (const float*)d_in[6];
  const float* lambdas = (const float*)d_in[7];
  float* out = (float*)d_out;
  char* ws = (char*)d_ws;

  const size_t OFF_PW   = 0;                               // 128*512*2
  const size_t OFF_P2W  = OFF_PW + (size_t)ND * NC * 2;    // 512*16384*2
  const size_t OFF_M    = OFF_P2W + (size_t)NF * DD * 2;   // 256*128*416*2
  const size_t OFF_B    = OFF_M + (size_t)NN * ND * SPAD * 2; // 256*128*128*4
  const size_t OFF_U    = OFF_B + (size_t)NN * DD * 4;
  const size_t OFF_W    = OFF_U + (size_t)NN * DD * 2;
  const size_t OFF_FLAT = OFF_W + (size_t)NN * DD * 2;

  _Float16* pw16  = (_Float16*)(ws + OFF_PW);
  _Float16* p2w16 = (_Float16*)(ws + OFF_P2W);
  _Float16* mf16  = (_Float16*)(ws + OFF_M);
  float*    Bws   = (float*)(ws + OFF_B);     // reused as Brec
  __bf16*   Uw    = (__bf16*)(ws + OFF_U);
  __bf16*   Ww    = (__bf16*)(ws + OFF_W);
  _Float16* flat  = (_Float16*)(ws + OFF_FLAT);

  k_cvt16<<<(ND * NC + 255) / 256, 256, 0, stream>>>(proj_w, pw16, ND * NC);
  k_cvt16<<<(NF * DD + 255) / 256, 256, 0, stream>>>(proj2_w, p2w16, NF * DD);
  k_roi_project<<<dim3(SPAD / 16, NN), 256, 0, stream>>>(images, bboxes, kf, pw16,
                                                         proj_b, mf16);
  k_gram<<<NN, 256, 0, stream>>>(mf16, Bws);
  k_jacobi<<<NN, 256, 0, stream>>>(Bws, lambdas, Uw, Ww);
  k_recon<<<NN, 256, 0, stream>>>(Ww, Uw, Bws);
  k_normflat<<<NN, 256, 0, stream>>>(Bws, flat);
  k_out<<<dim3(NF / 128, NN / 16), 256, 0, stream>>>(flat, p2w16, proj2_b, out);
}
